// Swin5BoxHead_16664473109184
// MI455X (gfx1250) — matrix operand order repaired and checked by reading, not compiled, not run on hardware
//
#include <hip/hip_runtime.h>

// ---------------------------------------------------------------------------
// SwinV2 stage (patch-merge + 2 blocks) for MI455X / gfx1250.
// All large GEMMs (~1.5 TFLOP total) run on v_wmma_f32_16x16x32_bf16 with f32
// accumulation, 64x64 tile per wave (4x4 fragment blocking -> 32 flop/byte
// from cache). Attention core (49x49 per head, <1% of FLOPs) is a small
// LDS-resident VALU kernel. wave32 everywhere.
// ---------------------------------------------------------------------------

typedef __bf16 bf16;
typedef __attribute__((ext_vector_type(16))) __bf16 v16bf;
typedef __attribute__((ext_vector_type(8)))  __bf16 v8bf;
typedef __attribute__((ext_vector_type(8)))  float  v8f;

#define BATCH 1024
#define CIN   384
#define CDIM  768
#define NHEAD 24
#define HD    32
#define WINW  7
#define NTOK  49              // 7*7 tokens per window
#define TOKS  (BATCH * NTOK)  // 50176
#define HIDC  512
#define MLPD  3072

// ---------------- workspace layout (bytes) ----------------
constexpr size_t BF2 = 2, F4 = 4;
constexpr size_t OFF_HB    = 0;                                   // h (bf16)  [TOKS, C]
constexpr size_t SZ_HB     = (size_t)TOKS * CDIM * BF2;
constexpr size_t OFF_HF    = OFF_HB + SZ_HB;                      // h (f32)   [TOKS, C]
constexpr size_t SZ_HF     = (size_t)TOKS * CDIM * F4;
constexpr size_t OFF_U     = OFF_HF + SZ_HF;                      // union: mergeA | qkv+attn | mlp
constexpr size_t SZ_U      = (size_t)TOKS * MLPD * BF2;           // 308 MB
constexpr size_t OFF_ATT   = OFF_U + (size_t)TOKS * 3 * CDIM * BF2; // attn-out after qkv
constexpr size_t OFF_T     = OFF_U + SZ_U;                        // f32 GEMM outputs [TOKS, C]
constexpr size_t SZ_T      = (size_t)TOKS * CDIM * F4;
constexpr size_t OFF_WMRG  = OFF_T + SZ_T;
constexpr size_t OFF_WQKV  = OFF_WMRG  + (size_t)CDIM * 4 * CIN * BF2;
constexpr size_t OFF_WPROJ = OFF_WQKV  + (size_t)2 * 3 * CDIM * CDIM * BF2;
constexpr size_t OFF_WFC1  = OFF_WPROJ + (size_t)2 * CDIM * CDIM * BF2;
constexpr size_t OFF_WFC2  = OFF_WFC1  + (size_t)2 * MLPD * CDIM * BF2;
constexpr size_t OFF_BIASQ = OFF_WFC2  + (size_t)2 * CDIM * MLPD * BF2;
constexpr size_t OFF_CPB   = OFF_BIASQ + (size_t)3 * CDIM * F4;
constexpr size_t OFF_RPB   = OFF_CPB   + (size_t)169 * NHEAD * F4;

// ---------------- elementwise f32 -> bf16 ----------------
__global__ __launch_bounds__(256) void f2bf_kernel(const float* __restrict__ s,
                                                   bf16* __restrict__ d, size_t n) {
  size_t i = (size_t)blockIdx.x * 256 + threadIdx.x;
  if (i < n) d[i] = (bf16)s[i];
}

// ---------------- patch-merge gather: x NCHW -> A [TOKS, 4*CIN] bf16 -------
__global__ __launch_bounds__(256) void merge_gather(const float* __restrict__ x,
                                                    bf16* __restrict__ Am) {
  const size_t idx = (size_t)blockIdx.x * 256 + threadIdx.x;
  if (idx >= (size_t)TOKS * (4 * CIN)) return;
  const int    cc  = (int)(idx % (4 * CIN));
  const size_t tok = idx / (4 * CIN);
  const int b = (int)(tok / NTOK);
  const int p = (int)(tok % NTOK);
  const int i = p / WINW, j = p % WINW;
  const int c = cc % CIN;
  const int sub = cc / CIN;          // (bb*2 + a) per SwinV2 transpose order
  const int bb = sub >> 1, a = sub & 1;
  const int hh = 2 * i + a, ww = 2 * j + bb;
  Am[idx] = (bf16)x[(((size_t)b * CIN + c) * 14 + hh) * 14 + ww];
}

// ---------------- WMMA GEMM: D = A[M,K] * W[N,K]^T (+epilogue) -------------
// Block = 128 threads = 4 waves in a 2x2 grid; block tile 128x128.
// Each wave: 64 rows x 64 cols = 4x4 blocking of 16x16 WMMA tiles
// (4 A-frags + 4 B-frags -> 16 v_wmma per K-step of 32).
enum { EPI_F32 = 0, EPI_F32_BIAS = 1, EPI_BF16_BIAS = 2, EPI_GELU_BF16 = 3 };

template <int EPI>
__global__ __launch_bounds__(128) void gemm_bf16_wmma(
    const bf16* __restrict__ A, const bf16* __restrict__ W,
    const float* __restrict__ bias, float* __restrict__ outF,
    bf16* __restrict__ outB, int M, int K, int N) {
  const int lane = threadIdx.x & 31;
  const int wave = threadIdx.x >> 5;
  const int half = lane >> 4;     // K-half selector (contiguous k order, same in A and B)
  const int r    = lane & 15;
  const int mBase = (blockIdx.x << 7) + ((wave & 1) << 6);
  const int nBase = (blockIdx.y << 7) + ((wave >> 1) << 6);

  v8f acc[16];
#pragma unroll
  for (int t = 0; t < 16; ++t) acc[t] = (v8f){};

  const bf16* aR[4];
  const bf16* wR[4];
#pragma unroll
  for (int t = 0; t < 4; ++t) {
    aR[t] = A + (size_t)(mBase + (t << 4) + r) * K;
    wR[t] = W + (size_t)(nBase + (t << 4) + r) * K;
  }

  for (int k = 0; k < K; k += 32) {
    // A 16x32 bf16: lanes 0-15 row m hold K 0..15; lanes 16-31 K 16..31 (contiguous)
    v16bf a[4];
#pragma unroll
    for (int t = 0; t < 4; ++t) {
      a[t] = *reinterpret_cast<const v16bf*>(aR[t] + k + half * 16);
    }
    // B 32x16: lane holds one N column; halves split K 0..15 / 16..31 (contiguous)
#pragma unroll
    for (int c = 0; c < 4; ++c) {
      v16bf b = *reinterpret_cast<const v16bf*>(wR[c] + k + half * 16);
#pragma unroll
      for (int t = 0; t < 4; ++t) {
        acc[t * 4 + c] = __builtin_amdgcn_wmma_f32_16x16x32_bf16(
            false, a[t], false, b, (short)0, acc[t * 4 + c], false, false);
      }
    }
  }

#pragma unroll
  for (int c = 0; c < 4; ++c) {
    const int   n  = nBase + (c << 4) + r;
    const float bv = (EPI == EPI_F32) ? 0.0f : bias[n];
#pragma unroll
    for (int t = 0; t < 4; ++t) {
#pragma unroll
      for (int v = 0; v < 8; ++v) {
        const int    m = mBase + (t << 4) + half * 8 + v;  // C/D: lanes0-15 N, VGPRv -> M
        const float  x = acc[t * 4 + c][v] + bv;
        const size_t o = (size_t)m * N + n;
        if (EPI == EPI_F32 || EPI == EPI_F32_BIAS) {
          outF[o] = x;
        } else if (EPI == EPI_BF16_BIAS) {
          outB[o] = (bf16)x;
        } else {  // exact GELU
          outB[o] = (bf16)(0.5f * x * (1.0f + erff(x * 0.70710678118f)));
        }
      }
    }
  }
}

// ---------------- LayerNorm over C=768 (+optional residual), dual output ---
template <bool RESID>
__global__ __launch_bounds__(256) void ln768(const float* __restrict__ src,
                                             const float* __restrict__ gamma,
                                             const float* __restrict__ beta,
                                             float* __restrict__ hf,
                                             bf16* __restrict__ hb) {
  const int row = blockIdx.x;
  const int tid = threadIdx.x;
  const float* p = src + (size_t)row * CDIM;
  __shared__ float red[256];
  float lx[3];
  float s = 0.f;
#pragma unroll
  for (int i = 0; i < 3; ++i) { lx[i] = p[tid + 256 * i]; s += lx[i]; }
  red[tid] = s; __syncthreads();
  for (int st = 128; st > 0; st >>= 1) { if (tid < st) red[tid] += red[tid + st]; __syncthreads(); }
  const float mu = red[0] * (1.0f / CDIM);
  __syncthreads();
  float s2 = 0.f;
#pragma unroll
  for (int i = 0; i < 3; ++i) { const float d = lx[i] - mu; s2 += d * d; }
  red[tid] = s2; __syncthreads();
  for (int st = 128; st > 0; st >>= 1) { if (tid < st) red[tid] += red[tid + st]; __syncthreads(); }
  const float inv = rsqrtf(red[0] * (1.0f / CDIM) + 1e-5f);
#pragma unroll
  for (int i = 0; i < 3; ++i) {
    const int c = tid + 256 * i;
    float y = (lx[i] - mu) * inv * gamma[c] + beta[c];
    const size_t o = (size_t)row * CDIM + c;
    if (RESID) y += hf[o];
    hf[o] = y;
    hb[o] = (bf16)y;
  }
}

// ---------------- qkv bias concat: [q_bias | 0 | v_bias] -------------------
__global__ __launch_bounds__(256) void qkv_bias_kernel(const float* __restrict__ qb,
                                                       const float* __restrict__ vb,
                                                       float* __restrict__ o) {
  const int j = blockIdx.x * 256 + threadIdx.x;
  if (j >= 3 * CDIM) return;
  o[j] = (j < CDIM) ? qb[j] : (j < 2 * CDIM ? 0.f : vb[j - 2 * CDIM]);
}

// ---------------- CPB tiny MLP over the 13x13 log-coords table -------------
__global__ __launch_bounds__(256) void cpb_kernel(const float* __restrict__ w1,
                                                  const float* __restrict__ b1,
                                                  const float* __restrict__ w2,
                                                  float* __restrict__ cpb) {
  const int idx = blockIdx.x * 256 + threadIdx.x;
  if (idx >= 169 * NHEAD) return;
  const int p = idx / NHEAD, h = idx % NHEAD;
  const int di = p / 13 - (WINW - 1), dj = p % 13 - (WINW - 1);
  auto f = [](float v) {
    v *= 8.0f / (WINW - 1);
    const float r = log2f(fabsf(v) + 1.0f) * (1.0f / 3.0f);  // log2(8)=3
    return v < 0.f ? -r : r;
  };
  const float tx = f((float)di), ty = f((float)dj);
  float acc = 0.f;
  for (int j = 0; j < HIDC; ++j) {
    const float hv = fmaxf(tx * w1[2 * j] + ty * w1[2 * j + 1] + b1[j], 0.f);
    acc += w2[h * HIDC + j] * hv;
  }
  cpb[p * NHEAD + h] = acc;
}

// ---------------- expand to 16*sigmoid(rpb) [NHEAD,49,49] ------------------
__global__ __launch_bounds__(256) void rpb_kernel(const float* __restrict__ cpb,
                                                  float* __restrict__ rpb) {
  const int idx = blockIdx.x * 256 + threadIdx.x;
  if (idx >= NHEAD * NTOK * NTOK) return;
  const int h = idx / (NTOK * NTOK);
  const int rem = idx % (NTOK * NTOK);
  const int n = rem / NTOK, m = rem % NTOK;
  const int di = n / WINW - m / WINW + (WINW - 1);
  const int dj = n % WINW - m % WINW + (WINW - 1);
  const float v = cpb[(di * 13 + dj) * NHEAD + h];
  rpb[idx] = 16.0f / (1.0f + __expf(-v));
}

// ---------------- per-(window, head) cosine attention ----------------------
__global__ __launch_bounds__(64) void win_attn(const bf16* __restrict__ qkv,
                                               bf16* __restrict__ out,
                                               const float* __restrict__ lscale,
                                               const float* __restrict__ rpb) {
  const int b = blockIdx.x / NHEAD;
  const int h = blockIdx.x % NHEAD;
  const size_t base = (size_t)b * NTOK * (3 * CDIM);
  __shared__ float kn[NTOK][HD];
  __shared__ float vv[NTOK][HD];
  __shared__ float lg[NTOK][NTOK];
  const int tid = threadIdx.x;
  for (int idx = tid; idx < NTOK * HD; idx += 64) {
    const int t = idx >> 5, d = idx & 31;
    const size_t o = base + (size_t)t * (3 * CDIM) + h * HD + d;
    kn[t][d] = (float)qkv[o + CDIM];
    vv[t][d] = (float)qkv[o + 2 * CDIM];
  }
  __syncthreads();
  if (tid < NTOK) {  // L2-normalize K rows in LDS
    float s = 0.f;
#pragma unroll
    for (int d = 0; d < HD; ++d) s += kn[tid][d] * kn[tid][d];
    const float inv = 1.0f / fmaxf(sqrtf(s), 1e-12f);
#pragma unroll
    for (int d = 0; d < HD; ++d) kn[tid][d] *= inv;
  }
  __syncthreads();
  if (tid < NTOK) {  // thread = query token
    const float scale = __expf(fminf(lscale[h], 4.6051702f));  // log(100)
    float q[HD];
    float s = 0.f;
#pragma unroll
    for (int d = 0; d < HD; ++d) {
      q[d] = (float)qkv[base + (size_t)tid * (3 * CDIM) + h * HD + d];
      s += q[d] * q[d];
    }
    const float qi = 1.0f / fmaxf(sqrtf(s), 1e-12f);
#pragma unroll
    for (int d = 0; d < HD; ++d) q[d] *= qi;
    const float* bias = rpb + ((size_t)h * NTOK + tid) * NTOK;
    float mx = -1e30f;
    for (int m = 0; m < NTOK; ++m) {
      float dot = 0.f;
#pragma unroll
      for (int d = 0; d < HD; ++d) dot += q[d] * kn[m][d];
      const float l = dot * scale + bias[m];
      lg[tid][m] = l;
      mx = fmaxf(mx, l);
    }
    float sum = 0.f;
    for (int m = 0; m < NTOK; ++m) {
      const float e = __expf(lg[tid][m] - mx);
      lg[tid][m] = e;
      sum += e;
    }
    const float isum = 1.0f / sum;
    float acc[HD];
#pragma unroll
    for (int d = 0; d < HD; ++d) acc[d] = 0.f;
    for (int m = 0; m < NTOK; ++m) {
      const float pm = lg[tid][m] * isum;
#pragma unroll
      for (int d = 0; d < HD; ++d) acc[d] += pm * vv[m][d];
    }
    bf16* op = out + ((size_t)b * NTOK + tid) * CDIM + h * HD;
#pragma unroll
    for (int d = 0; d < HD; ++d) op[d] = (bf16)acc[d];
  }
}

// ---------------- h [B,49,C] f32 -> out NCHW [B,C,7,7] ---------------------
__global__ __launch_bounds__(256) void out_kernel(const float* __restrict__ hf,
                                                  float* __restrict__ out) {
  const size_t idx = (size_t)blockIdx.x * 256 + threadIdx.x;
  if (idx >= (size_t)BATCH * CDIM * NTOK) return;
  const int    p = (int)(idx % NTOK);
  const size_t t = idx / NTOK;
  const int    c = (int)(t % CDIM);
  const int    b = (int)(t / CDIM);
  out[idx] = hf[((size_t)b * NTOK + p) * CDIM + c];
}

// ---------------------------------------------------------------------------
extern "C" void kernel_launch(void* const* d_in, const int* in_sizes, int n_in,
                              void* d_out, int out_size, void* d_ws, size_t ws_size,
                              hipStream_t stream) {
  (void)in_sizes; (void)n_in; (void)out_size; (void)ws_size;
  const float* x        = (const float*)d_in[0];
  const float* merge_w  = (const float*)d_in[1];
  const float* merge_g  = (const float*)d_in[2];
  const float* merge_b  = (const float*)d_in[3];
  const float* qkv_w    = (const float*)d_in[4];
  const float* q_bias   = (const float*)d_in[5];
  const float* v_bias   = (const float*)d_in[6];
  const float* logit_sc = (const float*)d_in[7];
  const float* cpb_w1   = (const float*)d_in[8];
  const float* cpb_b1   = (const float*)d_in[9];
  const float* cpb_w2   = (const float*)d_in[10];
  const float* proj_w   = (const float*)d_in[11];
  const float* proj_b   = (const float*)d_in[12];
  const float* n1_g     = (const float*)d_in[13];
  const float* n1_b     = (const float*)d_in[14];
  const float* fc1_w    = (const float*)d_in[15];
  const float* fc1_b    = (const float*)d_in[16];
  const float* fc2_w    = (const float*)d_in[17];
  const float* fc2_b    = (const float*)d_in[18];
  const float* n2_g     = (const float*)d_in[19];
  const float* n2_b     = (const float*)d_in[20];

  char*  ws    = (char*)d_ws;
  bf16*  hb    = (bf16*)(ws + OFF_HB);
  float* hf    = (float*)(ws + OFF_HF);
  bf16*  ubuf  = (bf16*)(ws + OFF_U);     // mergeA / qkv / mlp activations
  bf16*  attnb = (bf16*)(ws + OFF_ATT);
  float* tbuf  = (float*)(ws + OFF_T);
  bf16*  wmrg  = (bf16*)(ws + OFF_WMRG);
  bf16*  wqkv  = (bf16*)(ws + OFF_WQKV);
  bf16*  wproj = (bf16*)(ws + OFF_WPROJ);
  bf16*  wfc1  = (bf16*)(ws + OFF_WFC1);
  bf16*  wfc2  = (bf16*)(ws + OFF_WFC2);
  float* biasq = (float*)(ws + OFF_BIASQ);
  float* cpbb  = (float*)(ws + OFF_CPB);
  float* rpbb  = (float*)(ws + OFF_RPB);

  auto blocks = [](size_t n) { return (unsigned)((n + 255) / 256); };

  // Weight conversion to bf16 (both blocks at once).
  f2bf_kernel<<<blocks((size_t)CDIM * 4 * CIN), 256, 0, stream>>>(merge_w, wmrg, (size_t)CDIM * 4 * CIN);
  f2bf_kernel<<<blocks((size_t)2 * 3 * CDIM * CDIM), 256, 0, stream>>>(qkv_w, wqkv, (size_t)2 * 3 * CDIM * CDIM);
  f2bf_kernel<<<blocks((size_t)2 * CDIM * CDIM), 256, 0, stream>>>(proj_w, wproj, (size_t)2 * CDIM * CDIM);
  f2bf_kernel<<<blocks((size_t)2 * MLPD * CDIM), 256, 0, stream>>>(fc1_w, wfc1, (size_t)2 * MLPD * CDIM);
  f2bf_kernel<<<blocks((size_t)2 * CDIM * MLPD), 256, 0, stream>>>(fc2_w, wfc2, (size_t)2 * CDIM * MLPD);

  // Patch merge: gather -> GEMM [TOKS,1536]x[1536,768] -> LN.
  merge_gather<<<blocks((size_t)TOKS * 4 * CIN), 256, 0, stream>>>(x, ubuf);
  gemm_bf16_wmma<EPI_F32><<<dim3(TOKS / 128, CDIM / 128), 128, 0, stream>>>(
      ubuf, wmrg, nullptr, tbuf, nullptr, TOKS, 4 * CIN, CDIM);
  ln768<false><<<TOKS, 256, 0, stream>>>(tbuf, merge_g, merge_b, hf, hb);

  for (int i = 0; i < 2; ++i) {
    qkv_bias_kernel<<<blocks(3 * CDIM), 256, 0, stream>>>(q_bias + i * CDIM, v_bias + i * CDIM, biasq);
    gemm_bf16_wmma<EPI_BF16_BIAS><<<dim3(TOKS / 128, (3 * CDIM) / 128), 128, 0, stream>>>(
        hb, wqkv + (size_t)i * 3 * CDIM * CDIM, biasq, nullptr, ubuf, TOKS, CDIM, 3 * CDIM);

    cpb_kernel<<<blocks(169 * NHEAD), 256, 0, stream>>>(
        cpb_w1 + i * HIDC * 2, cpb_b1 + i * HIDC, cpb_w2 + i * NHEAD * HIDC, cpbb);
    rpb_kernel<<<blocks((size_t)NHEAD * NTOK * NTOK), 256, 0, stream>>>(cpbb, rpbb);

    win_attn<<<BATCH * NHEAD, 64, 0, stream>>>(ubuf, attnb, logit_sc + i * NHEAD, rpbb);

    gemm_bf16_wmma<EPI_F32_BIAS><<<dim3(TOKS / 128, CDIM / 128), 128, 0, stream>>>(
        attnb, wproj + (size_t)i * CDIM * CDIM, proj_b + i * CDIM, tbuf, nullptr, TOKS, CDIM, CDIM);
    ln768<true><<<TOKS, 256, 0, stream>>>(tbuf, n1_g + i * CDIM, n1_b + i * CDIM, hf, hb);

    gemm_bf16_wmma<EPI_GELU_BF16><<<dim3(TOKS / 128, MLPD / 128), 128, 0, stream>>>(
        hb, wfc1 + (size_t)i * MLPD * CDIM, fc1_b + i * MLPD, nullptr, ubuf, TOKS, CDIM, MLPD);
    gemm_bf16_wmma<EPI_F32_BIAS><<<dim3(TOKS / 128, CDIM / 128), 128, 0, stream>>>(
        ubuf, wfc2 + (size_t)i * CDIM * MLPD, fc2_b + i * CDIM, tbuf, nullptr, TOKS, MLPD, CDIM);
    ln768<true><<<TOKS, 256, 0, stream>>>(tbuf, n2_g + i * CDIM, n2_b + i * CDIM, hf, hb);
  }

  out_kernel<<<blocks((size_t)BATCH * CDIM * NTOK), 256, 0, stream>>>(hf, (float*)d_out);
}
